// MultiHeadAttention_10728828305867
// MI455X (gfx1250) — matrix operand order repaired and checked by reading, not compiled, not run on hardware
//
#include <hip/hip_runtime.h>

#define G  8
#define BB 16
#define NN 512
#define HH 512
#define DD 64

typedef __attribute__((ext_vector_type(16))) _Float16     v16h;
typedef __attribute__((ext_vector_type(8)))  float        v8f;
typedef __attribute__((ext_vector_type(4)))  unsigned int v4u;

union Frag { v16h h; v4u u[2]; };
union H2   { _Float16 h[2]; unsigned int u; };

// A-fragment (16x32, f16): row = lane&15; lanes<16 hold K = kOff+{0..7,16..23},
// lanes>=16 hold K = kOff+{8..15,24..31}. Two contiguous 16B loads per lane.
__device__ __forceinline__ v16h load_a_frag(const _Float16* base, int ld, int kOff,
                                            int lo, int hi) {
  Frag f;
  const _Float16* p = base + (size_t)lo * ld + kOff + hi * 8;
  f.u[0] = *(const v4u*)(p);
  f.u[1] = *(const v4u*)(p + 16);
  return f.h;
}

// B-fragment (32x16, f16) from B^T stored row-major (ld = contraction stride):
// col = lane&15; lanes<16 hold K = kOff+{0..7,16..23}, lanes>=16 hold K = kOff+{8..15,24..31}.
__device__ __forceinline__ v16h load_bt_frag(const _Float16* bt, int ld, int col,
                                             int kOff, int hi) {
  Frag f;
  const _Float16* p = bt + (size_t)col * ld + kOff + hi * 8;
  f.u[0] = *(const v4u*)(p);
  f.u[1] = *(const v4u*)(p + 16);
  return f.h;
}

__device__ __forceinline__ v8f wmma16(v16h a, v16h b, v8f c) {
  return __builtin_amdgcn_wmma_f32_16x16x32_f16(false, a, false, b, (short)0, c,
                                                false, false);
}

// ---------------- setup kernels ----------------
__global__ void k_cvt_f16(const float* __restrict__ src, _Float16* __restrict__ dst,
                          int n) {
  int i = blockIdx.x * blockDim.x + threadIdx.x;
  int stride = gridDim.x * blockDim.x;
  for (; i < n; i += stride) dst[i] = (_Float16)src[i];
}

// dst[s][c][r] = (f16) src[s][r][c]
__global__ void k_transpose_f16(const float* __restrict__ src,
                                _Float16* __restrict__ dst, int slabs, int R, int C) {
  int n = slabs * R * C;
  int i = blockIdx.x * blockDim.x + threadIdx.x;
  int stride = gridDim.x * blockDim.x;
  for (; i < n; i += stride) {
    int s = i / (R * C);
    int rem = i - s * (R * C);
    int c = rem / R;
    int r = rem - c * R;
    dst[i] = (_Float16)src[(size_t)s * R * C + (size_t)r * C + c];
  }
}

// ---------------- QKV projection ----------------
// wave = (g, b, 16-row strip). Q,K stored [g][b][n][d]; V stored transposed [g][b][d][n].
__global__ void __launch_bounds__(128) k_qkv(const _Float16* __restrict__ xh,
                                             const _Float16* __restrict__ wqt,
                                             const _Float16* __restrict__ wkt,
                                             const _Float16* __restrict__ wvt,
                                             _Float16* __restrict__ qh,
                                             _Float16* __restrict__ kh,
                                             _Float16* __restrict__ vt) {
  int wave = blockIdx.x * 4 + (threadIdx.x >> 5);
  int lane = threadIdx.x & 31;
  int lo = lane & 15, hi = lane >> 4;
  int g = wave >> 9;
  int b = (wave >> 5) & 15;
  int qt = wave & 31;

  const _Float16* xbase = xh + ((size_t)b * NN + qt * 16) * HH;
  const _Float16* wq = wqt + (size_t)g * DD * HH;
  const _Float16* wk = wkt + (size_t)g * DD * HH;
  const _Float16* wv = wvt + (size_t)g * DD * HH;

  v8f aq[4], ak[4], av[4];
#pragma unroll
  for (int d = 0; d < 4; ++d) { aq[d] = {}; ak[d] = {}; av[d] = {}; }

  for (int kt = 0; kt < HH / 32; ++kt) {
    int kOff = kt * 32;
    v16h A = load_a_frag(xbase, HH, kOff, lo, hi);
#pragma unroll
    for (int d = 0; d < 4; ++d) {
      v16h Bq = load_bt_frag(wq, HH, d * 16 + lo, kOff, hi);
      aq[d] = wmma16(A, Bq, aq[d]);
      v16h Bk = load_bt_frag(wk, HH, d * 16 + lo, kOff, hi);
      ak[d] = wmma16(A, Bk, ak[d]);
      v16h Bv = load_bt_frag(wv, HH, d * 16 + lo, kOff, hi);
      av[d] = wmma16(A, Bv, av[d]);
    }
  }

  size_t nbase = (size_t)(g * BB + b) * NN + qt * 16;
#pragma unroll
  for (int d = 0; d < 4; ++d) {
#pragma unroll
    for (int r = 0; r < 8; ++r) {
      size_t row = nbase + r + 8 * hi;
      qh[row * DD + d * 16 + lo] = (_Float16)aq[d][r];
      kh[row * DD + d * 16 + lo] = (_Float16)ak[d][r];
    }
    // V transposed: 8 consecutive rows per lane -> one packed 16B store
    unsigned int pu[4];
#pragma unroll
    for (int r = 0; r < 4; ++r) {
      H2 t;
      t.h[0] = (_Float16)av[d][2 * r];
      t.h[1] = (_Float16)av[d][2 * r + 1];
      pu[r] = t.u;
    }
    v4u vv = {pu[0], pu[1], pu[2], pu[3]};
    size_t vtoff = ((size_t)(g * BB + b) * DD + d * 16 + lo) * NN + qt * 16 + hi * 8;
    *(v4u*)(vt + vtoff) = vv;
  }
}

// ---------------- streaming attention ----------------
// wave = (g, b, 16 queries); iterate 32-key tiles with online softmax.
__global__ void __launch_bounds__(128) k_attn(const _Float16* __restrict__ qh,
                                              const _Float16* __restrict__ kh,
                                              const _Float16* __restrict__ vt,
                                              const float* __restrict__ edge,
                                              _Float16* __restrict__ oh) {
  __shared__ _Float16 pbuf[4][16 * 32];
  int wv = threadIdx.x >> 5;
  int wave = blockIdx.x * 4 + wv;
  int lane = threadIdx.x & 31;
  int lo = lane & 15, hi = lane >> 4;
  int g = wave >> 9;
  int b = (wave >> 5) & 15;
  int qt = wave & 31;

  const _Float16* qbase = qh + ((size_t)(g * BB + b) * NN + qt * 16) * DD;
  const _Float16* kbase = kh + (size_t)(g * BB + b) * NN * DD;
  const _Float16* vbase = vt + (size_t)(g * BB + b) * DD * NN;
  const float* ebase = edge + ((size_t)(g * BB + b) * NN + qt * 16) * NN;
  _Float16* pb = pbuf[wv];

  v16h Aq0 = load_a_frag(qbase, DD, 0, lo, hi);
  v16h Aq1 = load_a_frag(qbase, DD, 32, lo, hi);

  v8f accO[4];
#pragma unroll
  for (int d = 0; d < 4; ++d) accO[d] = {};
  float m[8], l[8];
#pragma unroll
  for (int r = 0; r < 8; ++r) { m[r] = -__builtin_inff(); l[r] = 0.f; }

  for (int kt = 0; kt < NN / 32; ++kt) {
    v8f s[2];
#pragma unroll
    for (int j = 0; j < 2; ++j) {
      v8f c = {};
      v16h B0 = load_bt_frag(kbase, DD, kt * 32 + j * 16 + lo, 0, hi);
      c = wmma16(Aq0, B0, c);
      v16h B1 = load_bt_frag(kbase, DD, kt * 32 + j * 16 + lo, 32, hi);
      c = wmma16(Aq1, B1, c);
      const float* ep = ebase + (size_t)(8 * hi) * NN + kt * 32 + j * 16 + lo;
#pragma unroll
      for (int r = 0; r < 8; ++r) c[r] = ep[(size_t)r * NN] + 0.125f * c[r];
      s[j] = c;
    }
    // per-row max over the 32 columns (cols live across lanes of each 16-group)
    float rmax[8];
#pragma unroll
    for (int r = 0; r < 8; ++r) rmax[r] = fmaxf(s[0][r], s[1][r]);
#pragma unroll
    for (int off = 8; off >= 1; off >>= 1)
#pragma unroll
      for (int r = 0; r < 8; ++r)
        rmax[r] = fmaxf(rmax[r], __shfl_xor(rmax[r], off, 16));
    float corr[8];
#pragma unroll
    for (int r = 0; r < 8; ++r) {
      float mn = fmaxf(m[r], rmax[r]);
      corr[r] = __expf(m[r] - mn);
      m[r] = mn;
    }
    float rsum[8];
#pragma unroll
    for (int r = 0; r < 8; ++r) {
      s[0][r] = __expf(s[0][r] - m[r]);
      s[1][r] = __expf(s[1][r] - m[r]);
      rsum[r] = s[0][r] + s[1][r];
    }
#pragma unroll
    for (int off = 8; off >= 1; off >>= 1)
#pragma unroll
      for (int r = 0; r < 8; ++r) rsum[r] += __shfl_xor(rsum[r], off, 16);
#pragma unroll
    for (int r = 0; r < 8; ++r) l[r] = l[r] * corr[r] + rsum[r];
#pragma unroll
    for (int d = 0; d < 4; ++d)
#pragma unroll
      for (int r = 0; r < 8; ++r) accO[d][r] *= corr[r];

    // C-fragment -> A-fragment layout change for P via per-wave LDS scratch
#pragma unroll
    for (int j = 0; j < 2; ++j)
#pragma unroll
      for (int r = 0; r < 8; ++r)
        pb[(r + 8 * hi) * 32 + j * 16 + lo] = (_Float16)s[j][r];
    asm volatile("s_wait_dscnt 0" ::: "memory");
    Frag fp;
    {
      const _Float16* p = pb + lo * 32 + hi * 8;
      fp.u[0] = *(const v4u*)(p);
      fp.u[1] = *(const v4u*)(p + 16);
    }
#pragma unroll
    for (int d = 0; d < 4; ++d) {
      v16h Bv = load_bt_frag(vbase, NN, d * 16 + lo, kt * 32, hi);
      accO[d] = wmma16(fp.h, Bv, accO[d]);
    }
    asm volatile("" ::: "memory");  // order LDS reuse across key tiles
  }

#pragma unroll
  for (int r = 0; r < 8; ++r) l[r] = 1.0f / l[r];
  size_t orow = (size_t)b * NN + qt * 16;
#pragma unroll
  for (int d = 0; d < 4; ++d)
#pragma unroll
    for (int r = 0; r < 8; ++r)
      oh[(orow + r + 8 * hi) * HH + g * DD + d * 16 + lo] =
          (_Float16)(accO[d][r] * l[r]);
}

// ---------------- output projection ----------------
// wave = (b, 16-row strip, 128-col strip)
__global__ void __launch_bounds__(128) k_proj(const _Float16* __restrict__ oh,
                                              const _Float16* __restrict__ wot,
                                              float* __restrict__ out) {
  int wave = blockIdx.x * 4 + (threadIdx.x >> 5);
  int lane = threadIdx.x & 31;
  int lo = lane & 15, hi = lane >> 4;
  int b = wave >> 7;
  int qt = (wave >> 2) & 31;
  int ct = wave & 3;

  const _Float16* obase = oh + ((size_t)b * NN + qt * 16) * HH;
  v8f acc[8];
#pragma unroll
  for (int j = 0; j < 8; ++j) acc[j] = {};
  for (int kt = 0; kt < HH / 32; ++kt) {
    int kOff = kt * 32;
    v16h A = load_a_frag(obase, HH, kOff, lo, hi);
#pragma unroll
    for (int j = 0; j < 8; ++j) {
      v16h Bw = load_bt_frag(wot, HH, ct * 128 + j * 16 + lo, kOff, hi);
      acc[j] = wmma16(A, Bw, acc[j]);
    }
  }
  size_t rbase = (size_t)b * NN + qt * 16;
#pragma unroll
  for (int j = 0; j < 8; ++j)
#pragma unroll
    for (int r = 0; r < 8; ++r)
      out[(rbase + r + 8 * hi) * HH + ct * 128 + j * 16 + lo] = acc[j][r];
}

extern "C" void kernel_launch(void* const* d_in, const int* in_sizes, int n_in,
                              void* d_out, int out_size, void* d_ws, size_t ws_size,
                              hipStream_t stream) {
  const float* x    = (const float*)d_in[0];
  const float* edge = (const float*)d_in[1];
  const float* Wq   = (const float*)d_in[2];
  const float* Wk   = (const float*)d_in[3];
  const float* Wv   = (const float*)d_in[4];
  const float* Wo   = (const float*)d_in[5];
  float* out = (float*)d_out;

  const size_t XN = (size_t)BB * NN * HH;      // 4,194,304 halves
  const size_t WN = (size_t)G * DD * HH;       //   262,144 halves
  const size_t QN = (size_t)G * BB * NN * DD;  // 4,194,304 halves

  _Float16* xh  = (_Float16*)d_ws;
  _Float16* wqt = xh + XN;
  _Float16* wkt = wqt + WN;
  _Float16* wvt = wkt + WN;
  _Float16* wot = wvt + WN;
  _Float16* qh  = wot + (size_t)HH * HH;
  _Float16* kh  = qh + QN;
  _Float16* vt  = kh + QN;
  _Float16* oh  = vt + QN;

  k_cvt_f16<<<2048, 256, 0, stream>>>(x, xh, (int)XN);
  k_transpose_f16<<<512, 256, 0, stream>>>(Wq, wqt, G, HH, DD);
  k_transpose_f16<<<512, 256, 0, stream>>>(Wk, wkt, G, HH, DD);
  k_transpose_f16<<<512, 256, 0, stream>>>(Wv, wvt, G, HH, DD);
  k_transpose_f16<<<512, 256, 0, stream>>>(Wo, wot, 1, HH, HH);

  k_qkv <<<1024, 128, 0, stream>>>(xh, wqt, wkt, wvt, qh, kh, vt);
  k_attn<<<1024, 128, 0, stream>>>(qh, kh, vt, edge, oh);
  k_proj<<< 512, 128, 0, stream>>>(oh, wot, out);
}
